// PointNet2ASIS_86500641341583
// MI455X (gfx1250) — compile-verified, matrix operand order checked by reading, not run on hardware
//
// PointNet++ ASIS forward for MI455X (gfx1250, wave32, WMMA).
// All matmuls (SA/FP MLPs, FC heads, ins_emb, sem_pred) go through one
// WMMA GEMM kernel: f32 in global, f16 staged in LDS (vectorized b128 loads
// on the block-uniform interior path; branch-free clamp+select on boundary),
// f32 accumulate via v_wmma_f32_16x16x32_f16. Working set ~90MB -> L2.
#include <hip/hip_runtime.h>

typedef __attribute__((ext_vector_type(16))) _Float16 v16h;
typedef __attribute__((ext_vector_type(8)))  float    v8f;
typedef __attribute__((ext_vector_type(4)))  _Float16 v4h;
typedef float v4f_u __attribute__((ext_vector_type(4), aligned(4))); // 4B-aligned vec load

#define TBM 128
#define TBN 64
#define TBK 32
#define KNNK 30

// ---------------------------------------------------------------------------
// WMMA GEMM + bias (+ optional gamma/beta affine + ReLU when mode==1)
//   Y[r, n] = f( X[r, :] @ W[:, n] + b[n] )
// Block = 256 threads = 8 waves; tile 128x64; each wave -> 32x32 (2x2 WMMA).
// ---------------------------------------------------------------------------
__global__ __launch_bounds__(256) void gemm_mlp_kernel(
    const float* __restrict__ X, const float* __restrict__ W,
    const float* __restrict__ bias, const float* __restrict__ gamma,
    const float* __restrict__ beta, float* __restrict__ Y,
    int rows, int K, int Co, int mode)
{
  __shared__ _Float16 Xs[TBM][TBK + 8];   // row stride 80B (8B aligned chunks)
  __shared__ _Float16 Ws[TBN][TBK + 8];
  const int tid  = threadIdx.x;
  const int wv   = tid >> 5;
  const int lane = tid & 31;
  const int wm   = wv >> 1;       // 0..3 : 32-row strip
  const int wn   = wv & 1;        // 0..1 : 32-col strip
  const int row0 = blockIdx.x * TBM;
  const int col0 = blockIdx.y * TBN;
  const int hl   = lane >> 4;     // lane half selects K sub-pattern
  const int lr   = lane & 15;

  const bool intM = (row0 + TBM <= rows);
  const bool intN = (col0 + TBN <= Co);

  v8f acc[2][2];
#pragma unroll
  for (int i = 0; i < 2; ++i)
#pragma unroll
    for (int j = 0; j < 2; ++j)
      acc[i][j] = (v8f){0.f, 0.f, 0.f, 0.f, 0.f, 0.f, 0.f, 0.f};

  for (int k0 = 0; k0 < K; k0 += TBK) {
    const bool fullK = (k0 + TBK <= K);

    // ---- stage X tile (128 x 32) as f16 ----
    if (intM && fullK) {
      // 1024 4-wide chunks, 4 per thread; straight-line, no divergence.
#pragma unroll
      for (int c = 0; c < 4; ++c) {
        int t = tid + c * 256;
        int r = t >> 3;             // 8 chunks per row
        int k4 = (t & 7) * 4;
        v4f_u v = *(const v4f_u*)(X + (size_t)(row0 + r) * K + (k0 + k4));
        *(v4h*)&Xs[r][k4] = __builtin_convertvector(v, v4h);
      }
      if (k0 + TBK < K)   // prefetch next K tile (global_prefetch_b8)
        __builtin_prefetch(X + (size_t)(row0 + (tid >> 3)) * K + (k0 + TBK), 0, 1);
    } else {
      // Boundary: clamp address (always legal) + select-zero; branch-free.
#pragma unroll
      for (int c = 0; c < 4; ++c) {
        int t = tid + c * 256;
        int r = t >> 3;
        int k4 = (t & 7) * 4;
        int gr = row0 + r;
        int rc = gr < rows ? gr : rows - 1;
        float vals[4];
#pragma unroll
        for (int u = 0; u < 4; ++u) {
          int gk = k0 + k4 + u;
          int kc = gk < K ? gk : K - 1;
          float v = X[(size_t)rc * K + kc];
          vals[u] = (gr < rows && gk < K) ? v : 0.f;
        }
        v4h h = {(_Float16)vals[0], (_Float16)vals[1],
                 (_Float16)vals[2], (_Float16)vals[3]};
        *(v4h*)&Xs[r][k4] = h;
      }
    }

    // ---- stage W tile (32 x 64) column-major as f16 ----
    if (intN && fullK) {
      // vectorize the *global* side over 4 contiguous columns (one b128),
      // scatter 4 b16 stores into the column-major LDS tile.
#pragma unroll
      for (int c = 0; c < 2; ++c) {
        int t = tid + c * 256;
        int k = t >> 4;             // 16 chunks per k-row
        int n4 = (t & 15) * 4;
        v4f_u v = *(const v4f_u*)(W + (size_t)(k0 + k) * Co + (col0 + n4));
        v4h h = __builtin_convertvector(v, v4h);
        Ws[n4 + 0][k] = h.x;
        Ws[n4 + 1][k] = h.y;
        Ws[n4 + 2][k] = h.z;
        Ws[n4 + 3][k] = h.w;
      }
      if (k0 + TBK < K)
        __builtin_prefetch(W + (size_t)(k0 + TBK + (tid >> 4)) * Co + col0, 0, 1);
    } else {
      // Boundary: clamp + select-zero; branch-free.
#pragma unroll
      for (int c = 0; c < 2; ++c) {
        int t = tid + c * 256;
        int k = t >> 4;
        int n4 = (t & 15) * 4;
        int gk = k0 + k;
        int kc = gk < K ? gk : K - 1;
        float vals[4];
#pragma unroll
        for (int u = 0; u < 4; ++u) {
          int gn = col0 + n4 + u;
          int nc = gn < Co ? gn : Co - 1;
          float v = W[(size_t)kc * Co + nc];
          vals[u] = (gk < K && gn < Co) ? v : 0.f;
        }
        Ws[n4 + 0][k] = (_Float16)vals[0];
        Ws[n4 + 1][k] = (_Float16)vals[1];
        Ws[n4 + 2][k] = (_Float16)vals[2];
        Ws[n4 + 3][k] = (_Float16)vals[3];
      }
    }
    __syncthreads();

    // CDNA5 16-bit A 16x32 layout: lane<16 -> K {0..7,16..23},
    // lane>=16 -> K {8..15,24..31}; B symmetric over columns.
    v16h af[2], bf[2];
#pragma unroll
    for (int i = 0; i < 2; ++i) {
      int mb = wm * 32 + i * 16;
      int nb = wn * 32 + i * 16;
#pragma unroll
      for (int e = 0; e < 16; ++e) {
        int vv = e >> 1, pp = e & 1;
        int kk = 2 * vv + pp + (vv >= 4 ? 8 : 0) + (hl ? 8 : 0);
        af[i][e] = Xs[mb + lr][kk];
        bf[i][e] = Ws[nb + lr][kk];
      }
    }
#pragma unroll
    for (int i = 0; i < 2; ++i)
#pragma unroll
      for (int j = 0; j < 2; ++j)
        acc[i][j] = __builtin_amdgcn_wmma_f32_16x16x32_f16(
            false, af[i], false, bf[j], (short)0, acc[i][j], false, false);
    __syncthreads();
  }

  // Epilogue: C/D layout VGPR e: lane<16 -> M=e, lane>=16 -> M=e+8.
  if (intM && intN) {
#pragma unroll
    for (int i = 0; i < 2; ++i)
#pragma unroll
      for (int j = 0; j < 2; ++j) {
        int nn = col0 + wn * 32 + j * 16 + lr;
        float bb = bias[nn];
        float gg = 1.f, ee = 0.f;
        if (mode) { gg = gamma[nn]; ee = beta[nn]; }
#pragma unroll
        for (int e = 0; e < 8; ++e) {
          int mm = row0 + wm * 32 + i * 16 + (hl ? 8 : 0) + e;
          float val = acc[i][j][e] + bb;
          if (mode) { val = fmaxf(val * gg + ee, 0.f); }
          Y[(size_t)mm * Co + nn] = val;
        }
      }
  } else {
#pragma unroll
    for (int i = 0; i < 2; ++i)
#pragma unroll
      for (int j = 0; j < 2; ++j) {
        int nn = col0 + wn * 32 + j * 16 + lr;
        int nc = nn < Co ? nn : Co - 1;
        float bb = bias[nc];
        float gg = 1.f, ee = 0.f;
        if (mode) { gg = gamma[nc]; ee = beta[nc]; }
#pragma unroll
        for (int e = 0; e < 8; ++e) {
          int mm = row0 + wm * 32 + i * 16 + (hl ? 8 : 0) + e;
          if (mm < rows && nn < Co) {
            float val = acc[i][j][e] + bb;
            if (mode) { val = fmaxf(val * gg + ee, 0.f); }
            Y[(size_t)mm * Co + nn] = val;
          }
        }
      }
  }
}

// ---------------------------------------------------------------------------
// x [B,9,N] -> xyz [B,N,3], pts [B,N,6]
// ---------------------------------------------------------------------------
__global__ void split_input_kernel(const float* __restrict__ x,
                                   float* __restrict__ xyz,
                                   float* __restrict__ pts, int n, int Btot)
{
  int t = blockIdx.x * blockDim.x + threadIdx.x;
  if (t >= Btot * n) return;
  int b = t / n, i = t % n;
  for (int c = 0; c < 3; ++c)
    xyz[((size_t)b * n + i) * 3 + c] = x[((size_t)b * 9 + c) * n + i];
  for (int c = 0; c < 6; ++c)
    pts[((size_t)b * n + i) * 6 + c] = x[((size_t)b * 9 + 3 + c) * n + i];
}

// ---------------------------------------------------------------------------
// Farthest-point sampling: one block per batch; dist cached in LDS.
// ---------------------------------------------------------------------------
__global__ __launch_bounds__(256) void fps_kernel(const float* __restrict__ xyz,
                                                  int n, int npoint,
                                                  int* __restrict__ out)
{
  __shared__ float dist[4096];
  __shared__ float rv[256];
  __shared__ int   ri[256];
  __shared__ int   s_last;
  const int b = blockIdx.x;
  const int tid = threadIdx.x;
  const float* p = xyz + (size_t)b * n * 3;
  for (int i = tid; i < n; i += 256) dist[i] = 1e10f;
  if (tid == 0) s_last = 0;
  __syncthreads();
  for (int it = 0; it < npoint; ++it) {
    int last = s_last;
    if (tid == 0) out[b * npoint + it] = last;
    float lx = p[last * 3 + 0], ly = p[last * 3 + 1], lz = p[last * 3 + 2];
    float bm = -1.f; int bi = 0;
    for (int i = tid; i < n; i += 256) {
      float dx = p[i * 3 + 0] - lx;
      float dy = p[i * 3 + 1] - ly;
      float dz = p[i * 3 + 2] - lz;
      float d = dx * dx + dy * dy + dz * dz;
      float nd = fminf(dist[i], d);
      dist[i] = nd;
      if (nd > bm) { bm = nd; bi = i; }
    }
    rv[tid] = bm; ri[tid] = bi;
    __syncthreads();
    for (int s = 128; s > 0; s >>= 1) {
      if (tid < s) {
        if (rv[tid + s] > rv[tid] ||
            (rv[tid + s] == rv[tid] && ri[tid + s] < ri[tid])) {
          rv[tid] = rv[tid + s]; ri[tid] = ri[tid + s];
        }
      }
      __syncthreads();
    }
    if (tid == 0) s_last = ri[0];
    __syncthreads();
  }
}

__global__ void gather_xyz_kernel(const float* __restrict__ xyz,
                                  const int* __restrict__ idx,
                                  float* __restrict__ nxyz, int n, int m, int Btot)
{
  int t = blockIdx.x * blockDim.x + threadIdx.x;
  if (t >= Btot * m) return;
  int b = t / m;
  int j = idx[t];
  for (int c = 0; c < 3; ++c)
    nxyz[(size_t)t * 3 + c] = xyz[((size_t)b * n + j) * 3 + c];
}

// Ball query: first `ns` indices (ascending) within radius; pad with first.
__global__ void ball_query_kernel(const float* __restrict__ xyz,
                                  const float* __restrict__ nxyz,
                                  int n, int m, float r2, int ns,
                                  int* __restrict__ idx, int Btot)
{
  int t = blockIdx.x * blockDim.x + threadIdx.x;
  if (t >= Btot * m) return;
  int b = t / m;
  const float* p = xyz + (size_t)b * n * 3;
  float qx = nxyz[(size_t)t * 3 + 0];
  float qy = nxyz[(size_t)t * 3 + 1];
  float qz = nxyz[(size_t)t * 3 + 2];
  int* o = idx + (size_t)t * ns;
  int cnt = 0, first = n - 1;
  for (int i = 0; i < n && cnt < ns; ++i) {
    float dx = p[i * 3 + 0] - qx;
    float dy = p[i * 3 + 1] - qy;
    float dz = p[i * 3 + 2] - qz;
    if (dx * dx + dy * dy + dz * dz <= r2) {
      if (cnt == 0) first = i;
      o[cnt++] = i;
    }
  }
  for (; cnt < ns; ++cnt) o[cnt] = first;
}

// feat[row] = concat(xyz[j]-new_xyz[q], pts[j])  ; row = ((b*m+q)*ns+s)
__global__ void sa_feat_kernel(const float* __restrict__ xyz,
                               const float* __restrict__ nxyz,
                               const float* __restrict__ pts,
                               const int* __restrict__ idx,
                               int n, int m, int ns, int C,
                               float* __restrict__ out, int Btot)
{
  int row = blockIdx.x * blockDim.x + threadIdx.x;
  if (row >= Btot * m * ns) return;
  int bm = row / ns;
  int b = bm / m;
  int j = idx[row];
  const float* pj = xyz + ((size_t)b * n + j) * 3;
  const float* nq = nxyz + (size_t)bm * 3;
  float* o = out + (size_t)row * (3 + C);
  o[0] = pj[0] - nq[0];
  o[1] = pj[1] - nq[1];
  o[2] = pj[2] - nq[2];
  const float* f = pts + ((size_t)b * n + j) * C;
  for (int c = 0; c < C; ++c) o[3 + c] = f[c];
}

__global__ void maxpool_kernel(const float* __restrict__ in,
                               float* __restrict__ out,
                               int groups, int ns, int C)
{
  int t = blockIdx.x * blockDim.x + threadIdx.x;
  if (t >= groups * C) return;
  int g = t / C, c = t % C;
  const float* p = in + ((size_t)g * ns) * C + c;
  float mx = -3.4e38f;
  for (int s = 0; s < ns; ++s) mx = fmaxf(mx, p[(size_t)s * C]);
  out[(size_t)g * C + c] = mx;
}

// FP: 3-NN inverse-distance interpolation, write concat(points1, interp).
__global__ void fp_feat_kernel(const float* __restrict__ xyz1,
                               const float* __restrict__ xyz2,
                               const float* __restrict__ pts1,
                               const float* __restrict__ pts2,
                               int n1, int n2, int C1, int C2,
                               float* __restrict__ out, int Btot)
{
  int t = blockIdx.x * blockDim.x + threadIdx.x;
  if (t >= Btot * n1) return;
  int b = t / n1;
  float qx = xyz1[(size_t)t * 3 + 0];
  float qy = xyz1[(size_t)t * 3 + 1];
  float qz = xyz1[(size_t)t * 3 + 2];
  const float* p2x = xyz2 + (size_t)b * n2 * 3;
  float d0 = 3.4e38f, d1 = 3.4e38f, d2 = 3.4e38f;
  int i0 = 0, i1 = 0, i2 = 0;
  for (int j = 0; j < n2; ++j) {
    float dx = p2x[j * 3 + 0] - qx;
    float dy = p2x[j * 3 + 1] - qy;
    float dz = p2x[j * 3 + 2] - qz;
    float d = dx * dx + dy * dy + dz * dz;
    if (d < d0)      { d2 = d1; i2 = i1; d1 = d0; i1 = i0; d0 = d; i0 = j; }
    else if (d < d1) { d2 = d1; i2 = i1; d1 = d; i1 = j; }
    else if (d < d2) { d2 = d; i2 = j; }
  }
  float w0 = 1.f / (fmaxf(d0, 0.f) + 1e-8f);
  float w1 = 1.f / (fmaxf(d1, 0.f) + 1e-8f);
  float w2 = 1.f / (fmaxf(d2, 0.f) + 1e-8f);
  float ws = w0 + w1 + w2;
  w0 /= ws; w1 /= ws; w2 /= ws;
  float* o = out + (size_t)t * (C1 + C2);
  const float* a = pts1 + (size_t)t * C1;
  for (int c = 0; c < C1; ++c) o[c] = a[c];
  const float* p2 = pts2 + (size_t)b * n2 * C2;
  for (int c = 0; c < C2; ++c)
    o[C1 + c] = w0 * p2[(size_t)i0 * C2 + c] +
                w1 * p2[(size_t)i1 * C2 + c] +
                w2 * p2[(size_t)i2 * C2 + c];
}

__global__ void add_kernel(const float* __restrict__ a,
                           const float* __restrict__ b,
                           float* __restrict__ o, int n)
{
  int t = blockIdx.x * blockDim.x + threadIdx.x;
  if (t < n) o[t] = a[t] + b[t];
}

// KNN over 5-dim embeddings: per query keep 30 smallest sq-distances.
__global__ void knn_kernel(const float* __restrict__ e, int n,
                           int* __restrict__ out, int Btot)
{
  int t = blockIdx.x * blockDim.x + threadIdx.x;
  if (t >= Btot * n) return;
  int b = t / n;
  const float* E = e + (size_t)b * n * 5;
  float q[5];
  for (int c = 0; c < 5; ++c) q[c] = e[(size_t)t * 5 + c];
  float bd[KNNK];
  int bi[KNNK];
  for (int k = 0; k < KNNK; ++k) { bd[k] = 3.4e38f; bi[k] = 0; }
  for (int j = 0; j < n; ++j) {
    float d = 0.f;
    for (int c = 0; c < 5; ++c) {
      float df = E[(size_t)j * 5 + c] - q[c];
      d += df * df;
    }
    if (d < bd[KNNK - 1]) {
      int p = KNNK - 1;
      while (p > 0 && bd[p - 1] > d) {
        bd[p] = bd[p - 1]; bi[p] = bi[p - 1]; --p;
      }
      bd[p] = d; bi[p] = j;
    }
  }
  int* o = out + (size_t)t * KNNK;
  for (int k = 0; k < KNNK; ++k) o[k] = bi[k];
}

// f_isem[b,i,c] = max_k f_sem[b, idx[b,i,k], c]
__global__ void isem_max_kernel(const float* __restrict__ fsem,
                                const int* __restrict__ idx,
                                int n, int C, float* __restrict__ out, int Btot)
{
  int t = blockIdx.x * blockDim.x + threadIdx.x;
  if (t >= Btot * n * C) return;
  int c = t % C;
  int bi_ = t / C;
  int b = bi_ / n;
  const float* base = fsem + (size_t)b * n * C;
  const int* id = idx + (size_t)bi_ * KNNK;
  float mx = -3.4e38f;
  for (int k = 0; k < KNNK; ++k)
    mx = fmaxf(mx, base[(size_t)id[k] * C + c]);
  out[t] = mx;
}

// [B,N,C] -> [B,C,N]
__global__ void transpose_bnc_bcn_kernel(const float* __restrict__ in,
                                         float* __restrict__ out,
                                         int n, int C, int Btot)
{
  int t = blockIdx.x * blockDim.x + threadIdx.x;
  if (t >= Btot * C * n) return;
  int b = t / (C * n);
  int r = t % (C * n);
  int c = r / n, i = r % n;
  out[t] = in[((size_t)b * n + i) * C + c];
}

// ---------------------------------------------------------------------------
static inline void launch_gemm(const float* X, const float* W, const float* b,
                               const float* g, const float* be, float* Y,
                               int rows, int K, int Co, int mode, hipStream_t s)
{
  dim3 grid((rows + TBM - 1) / TBM, (Co + TBN - 1) / TBN);
  gemm_mlp_kernel<<<grid, 256, 0, s>>>(X, W, b, g, be, Y, rows, K, Co, mode);
}

extern "C" void kernel_launch(void* const* d_in, const int* in_sizes, int n_in,
                              void* d_out, int out_size, void* d_ws, size_t ws_size,
                              hipStream_t stream)
{
  (void)in_sizes; (void)n_in; (void)out_size; (void)ws_size;
  const float* x = (const float*)d_in[0];
  auto F = [&](int i) -> const float* { return (const float*)d_in[i]; };

  // Flattened-param index map (insertion-order recursive flatten of the
  // reference dict; per layer: w,b,g,be):
  //  sa1@1 sa2@13 sa3@25 sa4@37 (3 layers x4 each)
  //  fp_sem4@49 fp_sem3@57 fp_sem2@65 fp_sem1@73 (2,2,2,3 layers)
  //  fp_ins4@85 fp_ins3@93 fp_ins2@101 fp_ins1@109
  //  sem_fc@121 ins_fc@125 adaptation@129  sem_pred w=133,b=134  ins_emb w=135,b=136
  const int Bc = 4, NS = 32, NCLS = 13, IDIM = 5;
  const int nl[5] = {4096, 1024, 256, 64, 16};
  const int Cl[5] = {6, 64, 128, 256, 512};
  const float rad2[4] = {0.01f, 0.04f, 0.16f, 0.64f};
  const int sa_base[4] = {1, 13, 25, 37};
  const int sa_cout[4][3] = {{32,32,64},{64,64,128},{128,128,256},{256,256,512}};
  const int fp_base[2][4] = {{73, 65, 57, 49}, {109, 101, 93, 85}}; // [br][lvl-1]
  const int fp_nlayers[4] = {3, 2, 2, 2};
  const int fp_couts[4][3] = {{128,128,128},{256,128,0},{256,256,0},{256,256,0}};

  // --- workspace allocator (offsets into d_ws, 256B aligned) ---
  size_t off = 0;
  char* ws = (char*)d_ws;
  auto alloc = [&](size_t bytes) -> void* {
    off = (off + 255) & ~(size_t)255;
    void* p = ws + off;
    off += bytes;
    return p;
  };
  float* xyzb[5]; float* psb[5];
  for (int l = 0; l < 5; ++l) {
    xyzb[l] = (float*)alloc((size_t)Bc * nl[l] * 3 * 4);
    psb[l]  = (float*)alloc((size_t)Bc * nl[l] * Cl[l] * 4);
  }
  int* fpsIdx  = (int*)alloc((size_t)Bc * 1024 * 4);
  int* ballIdx = (int*)alloc((size_t)Bc * 1024 * NS * 4);
  const size_t BIGF = (size_t)Bc * 1024 * NS * 64; // 8.39M floats
  float* featA   = (float*)alloc(BIGF * 4);
  float* featB   = (float*)alloc(BIGF * 4);
  float* semFeat = (float*)alloc((size_t)Bc * 4096 * 128 * 4);
  float* insFeat = (float*)alloc((size_t)Bc * 4096 * 128 * 4);
  float* eIns    = (float*)alloc((size_t)Bc * 4096 * IDIM * 4);
  int*   knnIdx  = (int*)alloc((size_t)Bc * 4096 * KNNK * 4);
  float* pSem    = (float*)alloc((size_t)Bc * 4096 * NCLS * 4);

  // --- 0) split input into xyz/pts (row-major per point) ---
  {
    int total = Bc * nl[0];
    split_input_kernel<<<(total + 255) / 256, 256, 0, stream>>>(
        x, xyzb[0], psb[0], nl[0], Bc);
  }

  // --- 1) set-abstraction levels ---
  for (int l = 0; l < 4; ++l) {
    int n = nl[l], m = nl[l + 1], Cin = Cl[l];
    fps_kernel<<<Bc, 256, 0, stream>>>(xyzb[l], n, m, fpsIdx);
    gather_xyz_kernel<<<(Bc * m + 255) / 256, 256, 0, stream>>>(
        xyzb[l], fpsIdx, xyzb[l + 1], n, m, Bc);
    ball_query_kernel<<<(Bc * m + 255) / 256, 256, 0, stream>>>(
        xyzb[l], xyzb[l + 1], n, m, rad2[l], NS, ballIdx, Bc);
    int rows = Bc * m * NS;
    sa_feat_kernel<<<(rows + 255) / 256, 256, 0, stream>>>(
        xyzb[l], xyzb[l + 1], psb[l], ballIdx, n, m, NS, Cin, featA, Bc);
    int cin = 3 + Cin;
    const float* src = featA; float* dst = featB;
    for (int j = 0; j < 3; ++j) {
      int base = sa_base[l] + 4 * j;
      int co = sa_cout[l][j];
      launch_gemm(src, F(base + 0), F(base + 1), F(base + 2), F(base + 3),
                  dst, rows, cin, co, 1, stream);
      cin = co;
      const float* tmp = src; src = dst; dst = (float*)tmp;
    }
    maxpool_kernel<<<(Bc * m * Cl[l + 1] + 255) / 256, 256, 0, stream>>>(
        src, psb[l + 1], Bc * m, NS, Cl[l + 1]);
  }

  // --- 2) feature-propagation branches (sem, ins) ---
  for (int br = 0; br < 2; ++br) {
    float* branchOut = (br == 0) ? semFeat : insFeat;
    const float* cur = psb[4];
    int curC = Cl[4];
    float* bufs2[2] = {featA, featB};
    for (int lvl = 4; lvl >= 1; --lvl) {
      const int n1 = nl[lvl - 1], n2 = nl[lvl];
      const int C1 = Cl[lvl - 1];
      const int rows = Bc * n1;
      float* cc = (cur == featA) ? featB : featA;
      fp_feat_kernel<<<(rows + 255) / 256, 256, 0, stream>>>(
          xyzb[lvl - 1], xyzb[lvl], psb[lvl - 1], cur, n1, n2, C1, curC, cc, Bc);
      int cin = C1 + curC;
      int six = (cc == featA) ? 0 : 1;
      const int L = fp_nlayers[lvl - 1];
      for (int j = 0; j < L; ++j) {
        const int base = fp_base[br][lvl - 1] + 4 * j;
        const int co = fp_couts[lvl - 1][j];
        const bool last = (lvl == 1 && j == L - 1);
        float* dst = last ? branchOut : bufs2[1 - six];
        launch_gemm(bufs2[six], F(base + 0), F(base + 1), F(base + 2),
                    F(base + 3), dst, rows, cin, co, 1, stream);
        cin = co;
        if (!last) six = 1 - six;
      }
      cur = (lvl == 1) ? branchOut : bufs2[six];
      curC = fp_couts[lvl - 1][L - 1];
    }
  }

  // --- 3) heads (featA/featB free now; carve head buffers out of them) ---
  const int rowsN = Bc * nl[0];
  float* fSem   = featA;
  float* fAdapt = featA + (size_t)rowsN * 128;
  float* fIsem  = featA + 2 * (size_t)rowsN * 128;
  float* fIns   = featB;
  float* fSins  = featB + (size_t)rowsN * 128;

  launch_gemm(semFeat, F(121), F(122), F(123), F(124), fSem,  rowsN, 128, 128, 1, stream);
  launch_gemm(insFeat, F(125), F(126), F(127), F(128), fIns,  rowsN, 128, 128, 1, stream);
  launch_gemm(fSem,    F(129), F(130), F(131), F(132), fAdapt, rowsN, 128, 128, 1, stream);
  add_kernel<<<(rowsN * 128 + 255) / 256, 256, 0, stream>>>(
      fIns, fAdapt, fSins, rowsN * 128);
  launch_gemm(fSins, F(135), F(136), nullptr, nullptr, eIns, rowsN, 128, IDIM, 0, stream);

  knn_kernel<<<(rowsN + 255) / 256, 256, 0, stream>>>(eIns, nl[0], knnIdx, Bc);
  isem_max_kernel<<<(rowsN * 128 + 255) / 256, 256, 0, stream>>>(
      fSem, knnIdx, nl[0], 128, fIsem, Bc);
  launch_gemm(fIsem, F(133), F(134), nullptr, nullptr, pSem, rowsN, 128, NCLS, 0, stream);

  // --- 4) outputs: [B,13,N] then [B,5,N] concatenated flat ---
  float* out = (float*)d_out;
  transpose_bnc_bcn_kernel<<<(Bc * NCLS * nl[0] + 255) / 256, 256, 0, stream>>>(
      pSem, out, nl[0], NCLS, Bc);
  transpose_bnc_bcn_kernel<<<(Bc * IDIM * nl[0] + 255) / 256, 256, 0, stream>>>(
      eIns, out + (size_t)Bc * NCLS * nl[0], nl[0], IDIM, Bc);
}
